// SeqAttentionWithPairBias_72627896975604
// MI455X (gfx1250) — compile-verified, matrix operand order checked by reading, not run on hardware
//
#include <hip/hip_runtime.h>
#include <hip/hip_bf16.h>

typedef __bf16 v16bf __attribute__((ext_vector_type(16)));
typedef float  v8f   __attribute__((ext_vector_type(8)));
typedef int    v4i   __attribute__((vector_size(16)));

#define WMMA_BF16(a, b, c) \
    __builtin_amdgcn_wmma_f32_16x16x32_bf16(false, (a), false, (b), (short)0, (c), false, false)

#if defined(__has_builtin)
#if __has_builtin(__builtin_amdgcn_global_load_async_to_lds_b128) && \
    __has_builtin(__builtin_amdgcn_s_wait_asynccnt)
#define HAVE_ASYNC_LDS 1
#endif
#endif
#ifndef HAVE_ASYNC_LDS
#define HAVE_ASYNC_LDS 0
#endif

// Native f32 -> bf16 (RNE): lowers to v_cvt_*bf16* on gfx1250.
__device__ __forceinline__ __bf16 tobf(float f) { return (__bf16)f; }
__device__ __forceinline__ __bf16 bfbits(unsigned short h) {
    return __builtin_bit_cast(__bf16, h);
}

// Load a 16x32 bf16 fragment (A layout; identical gather serves B when the
// source holds B^T row-major). lane&15 selects the M (or N) row; lane>>4
// selects the K half. VGPR j<4: K = kh*8 + 2j ; VGPR j>=4: K = 16 + kh*8 + 2(j-4).
__device__ __forceinline__ v16bf load_bf_frag(const __bf16* base, int ld,
                                              int row, int kbase, int lane) {
    v16bf f;
    int rn = lane & 15, kh = (lane >> 4) & 1;
    const __bf16* p = base + (size_t)(row + rn) * ld + kbase + kh * 8;
#pragma unroll
    for (int j = 0; j < 4; ++j) {
        unsigned v = *(const unsigned*)(p + 2 * j);
        f[2 * j]     = bfbits((unsigned short)(v & 0xffffu));
        f[2 * j + 1] = bfbits((unsigned short)(v >> 16));
    }
    const __bf16* p2 = p + 16;
#pragma unroll
    for (int j = 0; j < 4; ++j) {
        unsigned v = *(const unsigned*)(p2 + 2 * j);
        f[8 + 2 * j] = bfbits((unsigned short)(v & 0xffffu));
        f[9 + 2 * j] = bfbits((unsigned short)(v >> 16));
    }
    return f;
}

// Same fragment gather but from an f32 row-major source, converting to bf16.
__device__ __forceinline__ v16bf load_f32_frag(const float* base, int ld,
                                               int row, int kbase, int lane) {
    v16bf f;
    int rn = lane & 15, kh = (lane >> 4) & 1;
    const float* p = base + (size_t)(row + rn) * ld + kbase + kh * 8;
#pragma unroll
    for (int j = 0; j < 4; ++j) {
        f[2 * j]     = tobf(p[2 * j]);
        f[2 * j + 1] = tobf(p[2 * j + 1]);
    }
    const float* p2 = p + 16;
#pragma unroll
    for (int j = 0; j < 4; ++j) {
        f[8 + 2 * j] = tobf(p2[2 * j]);
        f[9 + 2 * j] = tobf(p2[2 * j + 1]);
    }
    return f;
}

// ---------------------------------------------------------------- LayerNorm(s)
__global__ __launch_bounds__(32)
void ln_s_kernel(const float* __restrict__ s, const float* __restrict__ w,
                 const float* __restrict__ b, __bf16* __restrict__ sn) {
    int row = blockIdx.x, lane = threadIdx.x;
    const float* x = s + (size_t)row * 384;
    float v[12], sum = 0.f, sq = 0.f;
#pragma unroll
    for (int i = 0; i < 12; ++i) {
        float t = x[lane + 32 * i];
        v[i] = t; sum += t; sq += t * t;
    }
#pragma unroll
    for (int m = 16; m >= 1; m >>= 1) {
        sum += __shfl_xor(sum, m, 32);
        sq  += __shfl_xor(sq, m, 32);
    }
    float mean = sum * (1.f / 384.f);
    float var  = sq * (1.f / 384.f) - mean * mean;
    float rstd = rsqrtf(var + 1e-5f);
#pragma unroll
    for (int i = 0; i < 12; ++i) {
        int c = lane + 32 * i;
        float y = (v[i] - mean) * rstd * w[c] + b[c];
        sn[(size_t)row * 384 + c] = tobf(y);
    }
}

// ------------------------------------------------- Q/K/V/G projections (WMMA)
__global__ __launch_bounds__(32)
void proj_kernel(const __bf16* __restrict__ sn,
                 const float* __restrict__ wq, const float* __restrict__ wk,
                 const float* __restrict__ wv, const float* __restrict__ wg,
                 const float* __restrict__ bg,
                 __bf16* __restrict__ qb, __bf16* __restrict__ kb,
                 __bf16* __restrict__ vt, float* __restrict__ gf) {
    int lane = threadIdx.x;
    int col0 = blockIdx.x * 16, row0 = blockIdx.y * 16;
    int which = blockIdx.z;
    const float* w = (which == 0) ? wq : (which == 1) ? wk : (which == 2) ? wv : wg;
    v8f acc = {};
    for (int k0 = 0; k0 < 384; k0 += 32) {
        v16bf a = load_bf_frag(sn, 384, row0, k0, lane);     // s_n rows
        v16bf bf = load_f32_frag(w, 384, col0, k0, lane);    // W rows = B^T rows
        acc = WMMA_BF16(a, bf, acc);
    }
    int n = lane & 15, mh = lane >> 4;
    int col = col0 + n;
    if (which == 0) {
        const float sc = 0.17677669529663687f;               // 32^-0.5
#pragma unroll
        for (int r = 0; r < 8; ++r) {
            int row = row0 + r + 8 * mh;
            qb[(size_t)row * 384 + col] = tobf(acc[r] * sc);
        }
    } else if (which == 1) {
#pragma unroll
        for (int r = 0; r < 8; ++r) {
            int row = row0 + r + 8 * mh;
            kb[(size_t)row * 384 + col] = tobf(acc[r]);
        }
    } else if (which == 2) {                                  // store V transposed
#pragma unroll
        for (int r = 0; r < 8; ++r) {
            int row = row0 + r + 8 * mh;
            vt[(size_t)col * 768 + row] = tobf(acc[r]);
        }
    } else {                                                  // gate: sigmoid(x+b)
        float bb = bg[col];
#pragma unroll
        for (int r = 0; r < 8; ++r) {
            int row = row0 + r + 8 * mh;
            float x = acc[r] + bb;
            gf[(size_t)row * 384 + col] = 1.0f / (1.0f + __expf(-x));
        }
    }
}

// ------------------------- fused LayerNorm(z) + pair-bias projection (stream z)
__global__ __launch_bounds__(32)
void bias_kernel(const float* __restrict__ z, const float* __restrict__ lw,
                 const float* __restrict__ lb, const float* __restrict__ wz,
                 float* __restrict__ bias) {
    __shared__ float zt[16 * 128];
    __shared__ float mn[16], rs[16];
    int lane = threadIdx.x;
    int j0 = blockIdx.x * 16;      // 16 consecutive j (keys)
    int i  = blockIdx.y;           // query row of z
    const float* src = z + ((size_t)i * 768 + j0) * 128;
#if HAVE_ASYNC_LDS
    // CDNA5 async memory->LDS DMA path (no VGPR round-trip, tracked by ASYNCcnt)
#pragma unroll
    for (int it = 0; it < 16; ++it) {
        int fi = it * 32 + lane;                    // 16B per lane per issue
        __builtin_amdgcn_global_load_async_to_lds_b128(
            (v4i*)(src + fi * 4), (v4i*)(zt + fi * 4), 0, 0);
    }
    __builtin_amdgcn_s_wait_asynccnt(0);
#else
#pragma unroll
    for (int it = 0; it < 16; ++it) {               // 16 rows x 128 f32, float4
        int fi = it * 32 + lane;
        ((float4*)zt)[fi] = ((const float4*)src)[fi];
    }
#endif
    __syncthreads();
    // per-row stats: 2 lanes per row, 64 elements each, combine with one shfl
    {
        int rrow = lane & 15, half = lane >> 4;
        const float4* rp = (const float4*)(zt + rrow * 128 + half * 64);
        float sum = 0.f, sq = 0.f;
#pragma unroll
        for (int c = 0; c < 16; ++c) {
            float4 t = rp[c];
            sum += t.x + t.y + t.z + t.w;
            sq  += t.x * t.x + t.y * t.y + t.z * t.z + t.w * t.w;
        }
        sum += __shfl_xor(sum, 16, 32);
        sq  += __shfl_xor(sq, 16, 32);
        if (lane < 16) {
            float mean = sum * (1.f / 128.f);
            float var  = sq * (1.f / 128.f) - mean * mean;
            mn[lane] = mean;
            rs[lane] = rsqrtf(var + 1e-5f);
        }
    }
    __syncthreads();
    int m = lane & 15, kh = lane >> 4;
    float mean = mn[m], rstd = rs[m];
    v8f acc = {};
    for (int s = 0; s < 4; ++s) {                     // K = 128 in 4 WMMA steps
        int kb = s * 32 + kh * 8;
        v16bf a, b;
#pragma unroll
        for (int j = 0; j < 8; ++j) {
            int c = kb + ((j < 4) ? 2 * j : 16 + 2 * (j - 4));
            float x0 = (zt[m * 128 + c]     - mean) * rstd * lw[c]     + lb[c];
            float x1 = (zt[m * 128 + c + 1] - mean) * rstd * lw[c + 1] + lb[c + 1];
            a[2 * j]     = tobf(x0);
            a[2 * j + 1] = tobf(x1);
            float w0 = (m < 12) ? wz[m * 128 + c]     : 0.f;   // B: n = lane&15 = head
            float w1 = (m < 12) ? wz[m * 128 + c + 1] : 0.f;
            b[2 * j]     = tobf(w0);
            b[2 * j + 1] = tobf(w1);
        }
        acc = WMMA_BF16(a, b, acc);
    }
    int n = lane & 15, mhh = lane >> 4;
    if (n < 12) {
#pragma unroll
        for (int r = 0; r < 8; ++r) {
            int jj = j0 + r + 8 * mhh;
            bias[((size_t)n * 768 + i) * 768 + jj] = acc[r];
        }
    }
}

// ------------------------------------- flash attention per (head, 16-query) tile
__global__ __launch_bounds__(32)
void attn_kernel(const __bf16* __restrict__ qb, const __bf16* __restrict__ kb,
                 const __bf16* __restrict__ vt, const float* __restrict__ bias,
                 const float* __restrict__ gf, __bf16* __restrict__ og) {
    __shared__ __bf16 pbuf[16 * 32];
    int lane = threadIdx.x;
    int q0 = blockIdx.x * 16;
    int h  = blockIdx.y;
    v16bf qf = load_bf_frag(qb, 384, q0, h * 32, lane);   // CH=32 == WMMA K
    float m_i[8], l_i[8];
    v8f o0 = {}, o1 = {};
#pragma unroll
    for (int r = 0; r < 8; ++r) { m_i[r] = -1e30f; l_i[r] = 0.f; }
    int n = lane & 15, mh = lane >> 4;
    const float* brow = bias + ((size_t)h * 768 + q0) * 768;
    for (int kc = 0; kc < 768; kc += 32) {
        v16bf b0 = load_bf_frag(kb, 384, kc,      h * 32, lane);  // keys 0-15
        v16bf b1 = load_bf_frag(kb, 384, kc + 16, h * 32, lane);  // keys 16-31
        v8f s0 = {}, s1 = {};
        s0 = WMMA_BF16(qf, b0, s0);
        s1 = WMMA_BF16(qf, b1, s1);
        float p0[8], p1[8];
#pragma unroll
        for (int r = 0; r < 8; ++r) {
            int m = r + 8 * mh;
            s0[r] += brow[(size_t)m * 768 + kc + n];
            s1[r] += brow[(size_t)m * 768 + kc + 16 + n];
            float mx = fmaxf(s0[r], s1[r]);
#pragma unroll
            for (int msk = 8; msk >= 1; msk >>= 1) mx = fmaxf(mx, __shfl_xor(mx, msk, 32));
            float mnew  = fmaxf(m_i[r], mx);
            float scale = __expf(m_i[r] - mnew);
            m_i[r] = mnew;
            l_i[r] *= scale;
            o0[r] *= scale;
            o1[r] *= scale;
            float e0 = __expf(s0[r] - mnew);
            float e1 = __expf(s1[r] - mnew);
            p0[r] = e0; p1[r] = e1;
            float rsum = e0 + e1;
#pragma unroll
            for (int msk = 8; msk >= 1; msk >>= 1) rsum += __shfl_xor(rsum, msk, 32);
            l_i[r] += rsum;
        }
        __syncthreads();                       // protect pbuf from previous pass
#pragma unroll
        for (int r = 0; r < 8; ++r) {          // C-layout -> row-major [m][k] in LDS
            int m = r + 8 * mh;
            pbuf[m * 32 + n]      = tobf(p0[r]);
            pbuf[m * 32 + 16 + n] = tobf(p1[r]);
        }
        __syncthreads();
        v16bf pa  = load_bf_frag(pbuf, 32, 0, 0, lane);           // A layout
        v16bf vb0 = load_bf_frag(vt, 768, h * 32,      kc, lane); // ch 0-15
        v16bf vb1 = load_bf_frag(vt, 768, h * 32 + 16, kc, lane); // ch 16-31
        o0 = WMMA_BF16(pa, vb0, o0);
        o1 = WMMA_BF16(pa, vb1, o1);
    }
#pragma unroll
    for (int r = 0; r < 8; ++r) {
        int row = q0 + r + 8 * mh;
        float inv = 1.0f / l_i[r];
        int c0 = h * 32 + n, c1 = c0 + 16;
        float g0 = gf[(size_t)row * 384 + c0];
        float g1 = gf[(size_t)row * 384 + c1];
        og[(size_t)row * 384 + c0] = tobf(o0[r] * inv * g0);
        og[(size_t)row * 384 + c1] = tobf(o1[r] * inv * g1);
    }
}

// ------------------------------------------------------ output projection (WMMA)
__global__ __launch_bounds__(32)
void out_kernel(const __bf16* __restrict__ og, const float* __restrict__ wo,
                const float* __restrict__ bo, float* __restrict__ out) {
    int lane = threadIdx.x;
    int col0 = blockIdx.x * 16, row0 = blockIdx.y * 16;
    v8f acc = {};
    for (int k0 = 0; k0 < 384; k0 += 32) {
        v16bf a = load_bf_frag(og, 384, row0, k0, lane);
        v16bf b = load_f32_frag(wo, 384, col0, k0, lane);   // w_o[CS, H*CH]
        acc = WMMA_BF16(a, b, acc);
    }
    int n = lane & 15, mh = lane >> 4;
    int col = col0 + n;
    float bb = bo[col];
#pragma unroll
    for (int r = 0; r < 8; ++r) {
        int row = row0 + r + 8 * mh;
        out[(size_t)row * 384 + col] = acc[r] + bb;
    }
}

extern "C" void kernel_launch(void* const* d_in, const int* in_sizes, int n_in,
                              void* d_out, int out_size, void* d_ws, size_t ws_size,
                              hipStream_t stream) {
    (void)in_sizes; (void)n_in; (void)out_size; (void)ws_size;
    const float* s      = (const float*)d_in[0];
    const float* z      = (const float*)d_in[1];
    const float* ln_s_w = (const float*)d_in[2];
    const float* ln_s_b = (const float*)d_in[3];
    const float* ln_z_w = (const float*)d_in[4];
    const float* ln_z_b = (const float*)d_in[5];
    const float* w_z    = (const float*)d_in[6];
    const float* w_q    = (const float*)d_in[7];
    const float* w_k    = (const float*)d_in[8];
    const float* w_v    = (const float*)d_in[9];
    const float* w_g    = (const float*)d_in[10];
    const float* b_g    = (const float*)d_in[11];
    const float* w_o    = (const float*)d_in[12];
    const float* b_o    = (const float*)d_in[13];
    float* out = (float*)d_out;

    char* ws = (char*)d_ws;
    __bf16* sn   = (__bf16*)(ws + 0);          // 768*384 bf16  = 589824 B
    __bf16* qb   = (__bf16*)(ws + 589824);     // 589824 B
    __bf16* kb   = (__bf16*)(ws + 1179648);    // 589824 B
    __bf16* vt   = (__bf16*)(ws + 1769472);    // V transposed [384,768], 589824 B
    float*  gf   = (float*)(ws + 2359296);     // gate f32, 1179648 B
    __bf16* og   = (__bf16*)(ws + 3538944);    // gated attn out, 589824 B
    float*  bias = (float*)(ws + 4128768);     // [12,768,768] f32, 28311552 B

    ln_s_kernel<<<768, 32, 0, stream>>>(s, ln_s_w, ln_s_b, sn);
    proj_kernel<<<dim3(24, 48, 4), 32, 0, stream>>>(sn, w_q, w_k, w_v, w_g, b_g,
                                                    qb, kb, vt, gf);
    bias_kernel<<<dim3(48, 768), 32, 0, stream>>>(z, ln_z_w, ln_z_b, w_z, bias);
    attn_kernel<<<dim3(48, 12), 32, 0, stream>>>(qb, kb, vt, bias, gf, og);
    out_kernel<<<dim3(24, 48), 32, 0, stream>>>(og, w_o, b_o, out);
}
